// MultiHeadAttention_14422500180134
// MI455X (gfx1250) — matrix operand order repaired and checked by reading, not compiled, not run on hardware
//
#include <hip/hip_runtime.h>

#define B_ 2
#define S_ 2048
#define D_ 1024
#define H_ 16
#define DK_ 64

typedef __bf16 bf16_t;
typedef __attribute__((ext_vector_type(16))) __bf16 v16bf;
typedef __attribute__((ext_vector_type(8)))  float  v8f;

// ---- helpers ---------------------------------------------------------------

__device__ __forceinline__ bf16_t f2bf(float f) {
  union { float f; unsigned u; } in; in.f = f;
  unsigned u = in.u;
  u += 0x7FFFu + ((u >> 16) & 1u);              // round-to-nearest-even
  union { unsigned short s; bf16_t b; } out;
  out.s = (unsigned short)(u >> 16);
  return out.b;
}

// Low 32 bits of a generic pointer into LDS == byte offset in the wave's LDS
// allocation (flat LDS aperture truncates to addr[31:0]).
__device__ __forceinline__ unsigned lds_off(const void* p) {
  return (unsigned)(unsigned long long)p;
}

// 16-byte async global->LDS copy (ASYNCcnt-tracked).
__device__ __forceinline__ void async_ld16(unsigned ldsByteOff, const void* gaddr) {
  asm volatile("global_load_async_to_lds_b128 %0, %1, off"
               :: "v"(ldsByteOff), "v"((unsigned long long)gaddr)
               : "memory");
}
__device__ __forceinline__ void wait_async0() {
  asm volatile("s_wait_asynccnt 0" ::: "memory");
}

// A-fragment, 16x32 bf16 (M x K). Row-major source, ld in elements.
__device__ __forceinline__ v16bf load_frag_a(const bf16_t* base, int ld,
                                             int m0, int k0, int lane) {
  int m  = m0 + (lane & 15);
  int kb = k0 + ((lane & 16) >> 1);             // 0 or 8
  const bf16_t* p = base + (size_t)m * ld + kb;
  union { v16bf v; uint4 q[2]; } u;
  u.q[0] = *(const uint4*)(p);                  // K = kb .. kb+7
  u.q[1] = *(const uint4*)(p + 16);             // K = kb+16 .. kb+23
  return u.v;
}

// B-fragment, 32x16 bf16 (K x N), taken from row-major Bt[N][K] (NT form).
__device__ __forceinline__ v16bf load_frag_b(const bf16_t* baseT, int ld,
                                             int n0, int k0, int lane) {
  int n  = n0 + (lane & 15);
  int kb = k0 + ((lane & 16) >> 1);             // 0 or 8
  const bf16_t* p = baseT + (size_t)n * ld + kb;
  union { v16bf v; uint4 q[2]; } u;
  u.q[0] = *(const uint4*)(p);
  u.q[1] = *(const uint4*)(p + 16);
  return u.v;
}

__device__ __forceinline__ v8f wmma_bf16f32(v16bf a, v16bf b, v8f c) {
  return __builtin_amdgcn_wmma_f32_16x16x32_bf16(
      false, a, false, b, (short)0, c, false, false);
}

// ---- kernel 1: f32 -> bf16 -------------------------------------------------

__global__ void k_cvt_bf16(const float* __restrict__ src, bf16_t* __restrict__ dst, int n) {
  int i = blockIdx.x * blockDim.x + threadIdx.x;
  if (i < n) dst[i] = f2bf(src[i]);
}

// ---- kernel 2: Wt[n][k] = bf16(W[k][n]) ------------------------------------

__global__ void k_transpose_bf16(const float* __restrict__ W, bf16_t* __restrict__ Wt,
                                 int K, int N) {
  int i = blockIdx.x * blockDim.x + threadIdx.x;
  if (i >= K * N) return;
  int n = i / K, k = i - n * K;
  Wt[(size_t)n * K + k] = f2bf(W[(size_t)k * N + n]);
}

// ---- kernel 3: QKV projection GEMM-NT --------------------------------------
// C[4096,1024] = X[4096,1024] x Wt[1024,1024]^T + bias
// A-stripe (16 rows x full K = contiguous 32KB) async-staged into LDS once per
// block; all 8 waves share it. B streamed from global (L2-resident).

__global__ void k_qkv_proj(const bf16_t* __restrict__ X, const bf16_t* __restrict__ Wt,
                           const float* __restrict__ bias, bf16_t* __restrict__ dst,
                           int vt_mode) {
  __shared__ bf16_t As[16 * D_];                // 32 KB
  int wave = threadIdx.x >> 5, lane = threadIdx.x & 31;
  int m0 = (((blockIdx.x * 8) >> 6) << 4);      // uniform across the block
  int n0 = ((blockIdx.x * 8 + wave) & 63) << 4;

  {
    const char* src = (const char*)(X + (size_t)m0 * D_);
    unsigned base = lds_off(&As[0]);
#pragma unroll
    for (int i = 0; i < 8; ++i) {               // 256 thr x 128 B = 32 KB
      unsigned off = threadIdx.x * 128 + i * 16;
      async_ld16(base + off, src + off);
    }
    wait_async0();
  }
  __syncthreads();

  v8f acc = {};
  for (int k0 = 0; k0 < D_; k0 += 32) {
    v16bf a = load_frag_a(As, D_, 0, k0, lane);
    v16bf b = load_frag_b(Wt, D_, n0, k0, lane);
    acc = wmma_bf16f32(a, b, acc);
  }
  int n  = n0 + (lane & 15);
  int h  = n >> 6, dk = n & (DK_ - 1);
  float bi = bias[n];
#pragma unroll
  for (int r = 0; r < 8; ++r) {
    int m  = m0 + r + ((lane & 16) >> 1);
    int bb = m >> 11, s = m & (S_ - 1);
    float v = acc[r] + bi;
    if (vt_mode == 0)
      dst[(((size_t)bb * H_ + h) * S_ + s) * DK_ + dk] = f2bf(v);
    else
      dst[(((size_t)bb * H_ + h) * DK_ + dk) * S_ + s] = f2bf(v);
  }
}

// ---- kernel 4: fused attention ---------------------------------------------
// One workgroup = (b, h, 16 q-rows). ~200 KB dynamic LDS (320 KB WGP budget):
//   sc   : 16 x 2048 f32 scores         (128 KB)
//   pbf  : 16 x 2048 bf16 probabilities ( 64 KB)
//   red  : 288 f32 reduction scratch
//   oacc : 16 x 64 f32 P*V accumulator  (  4 KB)
//   qs   : 16 x 64 bf16 Q tile          (  2 KB, async-staged)

__global__ void k_attn(const bf16_t* __restrict__ Qh, const bf16_t* __restrict__ Kh,
                       const bf16_t* __restrict__ Vt, const unsigned char* __restrict__ mask,
                       float* __restrict__ attn, bf16_t* __restrict__ O) {
  extern __shared__ char smem[];
  float*  sc   = (float*)smem;                                   // 16*2048
  bf16_t* pbf  = (bf16_t*)(smem + 16 * S_ * 4);                  // 16*2048
  float*  red  = (float*)(smem + 16 * S_ * 4 + 16 * S_ * 2);     // 288
  float*  oacc = red + 288;                                      // 16*64
  bf16_t* qs   = (bf16_t*)(oacc + 16 * 64);                      // 16*64

  int bh = blockIdx.x >> 7;                     // S_/16 = 128 q-tiles
  int q0 = (blockIdx.x & 127) << 4;
  int b  = bh >> 4, h = bh & 15;
  int wave = threadIdx.x >> 5, lane = threadIdx.x & 31;

  const bf16_t* Qb = Qh + ((size_t)bh * S_ + q0) * DK_;  // 16 x 64 contiguous
  const bf16_t* Kb = Kh + (size_t)bh * S_ * DK_;         // 2048 x 64, ld 64

  // async-stage the 2 KB Q tile (waves 0-3, 16 B per lane)
  if (threadIdx.x < 128) {
    unsigned off = threadIdx.x * 16;
    async_ld16(lds_off(qs) + off, (const char*)Qb + off);
    wait_async0();
  }
  __syncthreads();

  // -- phase 1: scores = (Q K^T) / sqrt(DK), masked, into LDS --
  v16bf a0 = load_frag_a(qs, DK_, 0, 0, lane);
  v16bf a1 = load_frag_a(qs, DK_, 0, 32, lane);
  for (int kt = wave; kt < S_ / 16; kt += 8) {
    v8f acc = {};
    v16bf b0 = load_frag_b(Kb, DK_, kt * 16, 0, lane);
    acc = wmma_bf16f32(a0, b0, acc);
    v16bf b1 = load_frag_b(Kb, DK_, kt * 16, 32, lane);
    acc = wmma_bf16f32(a1, b1, acc);
    int c = kt * 16 + (lane & 15);
#pragma unroll
    for (int r = 0; r < 8; ++r) {
      int row = r + ((lane & 16) >> 1);
      float v = acc[r] * 0.125f;                // 1/sqrt(64)
      unsigned char mk =
          __builtin_nontemporal_load(&mask[((size_t)b * S_ + (q0 + row)) * S_ + c]);
      sc[row * S_ + c] = mk ? v : -3.4028235e38f;
    }
  }
  __syncthreads();

  // -- phase 2: row softmax in LDS; emit attn (f32, NT stores) + P (bf16, LDS)
  {
    int row = threadIdx.x >> 4, g = threadIdx.x & 15;
    float* srow = sc + row * S_;
    int c0 = g * 128;
    float lmax = -3.4028235e38f;
    for (int c = c0; c < c0 + 128; ++c) lmax = fmaxf(lmax, srow[c]);
    red[row * 16 + g] = lmax;
    __syncthreads();
    if (g == 0) {
      float m = red[row * 16];
      for (int j = 1; j < 16; ++j) m = fmaxf(m, red[row * 16 + j]);
      red[256 + row] = m;
    }
    __syncthreads();
    float rmax = red[256 + row];
    float lsum = 0.f;
    for (int c = c0; c < c0 + 128; ++c) {
      float e = __expf(srow[c] - rmax);
      srow[c] = e;
      lsum += e;
    }
    red[row * 16 + g] = lsum;
    __syncthreads();
    if (g == 0) {
      float s = 0.f;
      for (int j = 0; j < 16; ++j) s += red[row * 16 + j];
      red[272 + row] = s;
    }
    __syncthreads();
    float inv = 1.f / red[272 + row];
    size_t abase = ((size_t)bh * S_ + (q0 + row)) * (size_t)S_;
    for (int c = c0; c < c0 + 128; ++c) {
      float p = srow[c] * inv;
      __builtin_nontemporal_store(p, &attn[abase + c]);   // write-once stream
      pbf[row * S_ + c] = f2bf(p);
    }
  }

  for (int i = threadIdx.x; i < 16 * 64; i += 256) oacc[i] = 0.f;
  __syncthreads();

  // -- phase 3: O_tile = P x V. 8 waves = 4 dk-tiles x 2 k-halves; LDS add --
  {
    int dkt = wave & 3, kh = wave >> 2;
    const bf16_t* Vb = Vt + (size_t)bh * DK_ * S_;     // rows dk, ld S_
    v8f acc = {};
    for (int k0 = kh * 1024; k0 < kh * 1024 + 1024; k0 += 32) {
      v16bf a  = load_frag_a(pbf, S_, 0, k0, lane);    // P from LDS
      v16bf bf = load_frag_b(Vb,  S_, dkt * 16, k0, lane);
      acc = wmma_bf16f32(a, bf, acc);
    }
#pragma unroll
    for (int r = 0; r < 8; ++r) {
      int row = r + ((lane & 16) >> 1);
      int cc  = dkt * 16 + (lane & 15);
      atomicAdd(&oacc[row * 64 + cc], acc[r]);         // ds_add_f32
    }
  }
  __syncthreads();

  for (int i = threadIdx.x; i < 16 * 64; i += 256) {
    int row = i >> 6, cc = i & 63;
    O[((size_t)b * S_ + (q0 + row)) * D_ + h * DK_ + cc] = f2bf(oacc[i]);
  }
}

// ---- kernel 5: output projection -------------------------------------------
// out[4096,1024] = O[4096,1024] x WoT[1024,1024]^T + bo   (f32 out)

__global__ void k_oproj(const bf16_t* __restrict__ O, const bf16_t* __restrict__ WoT,
                        const float* __restrict__ bo, float* __restrict__ out) {
  __shared__ bf16_t As[16 * D_];                // 32 KB
  int wave = threadIdx.x >> 5, lane = threadIdx.x & 31;
  int m0 = (((blockIdx.x * 8) >> 6) << 4);
  int n0 = ((blockIdx.x * 8 + wave) & 63) << 4;

  {
    const char* src = (const char*)(O + (size_t)m0 * D_);
    unsigned base = lds_off(&As[0]);
#pragma unroll
    for (int i = 0; i < 8; ++i) {
      unsigned off = threadIdx.x * 128 + i * 16;
      async_ld16(base + off, src + off);
    }
    wait_async0();
  }
  __syncthreads();

  v8f acc = {};
  for (int k0 = 0; k0 < D_; k0 += 32) {
    v16bf a = load_frag_a(As, D_, 0, k0, lane);
    v16bf b = load_frag_b(WoT, D_, n0, k0, lane);
    acc = wmma_bf16f32(a, b, acc);
  }
  int n = n0 + (lane & 15);
  float bi = bo[n];
#pragma unroll
  for (int r = 0; r < 8; ++r) {
    int m = m0 + r + ((lane & 16) >> 1);
    out[(size_t)m * D_ + n] = acc[r] + bi;
  }
}

// ---- host ------------------------------------------------------------------

extern "C" void kernel_launch(void* const* d_in, const int* in_sizes, int n_in,
                              void* d_out, int out_size, void* d_ws, size_t ws_size,
                              hipStream_t stream) {
  const float*         x    = (const float*)d_in[0];
  const unsigned char* mask = (const unsigned char*)d_in[1];
  const float* Wq = (const float*)d_in[2];
  const float* bq = (const float*)d_in[3];
  const float* Wk = (const float*)d_in[4];
  const float* bk = (const float*)d_in[5];
  const float* Wv = (const float*)d_in[6];
  const float* bv = (const float*)d_in[7];
  const float* Wo = (const float*)d_in[8];
  const float* bo = (const float*)d_in[9];

  char* ws = (char*)d_ws;                       // needs >= 48 MB
  bf16_t* Xbf = (bf16_t*)(ws);
  bf16_t* WqT = (bf16_t*)(ws + (size_t)( 8 << 20));
  bf16_t* WkT = (bf16_t*)(ws + (size_t)(10 << 20));
  bf16_t* WvT = (bf16_t*)(ws + (size_t)(12 << 20));
  bf16_t* WoT = (bf16_t*)(ws + (size_t)(14 << 20));
  bf16_t* Qh  = (bf16_t*)(ws + (size_t)(16 << 20));  // [B,H,S,DK]
  bf16_t* Kh  = (bf16_t*)(ws + (size_t)(24 << 20));  // [B,H,S,DK]
  bf16_t* Vt  = (bf16_t*)(ws + (size_t)(32 << 20));  // [B,H,DK,S]
  bf16_t* O   = (bf16_t*)(ws + (size_t)(40 << 20));  // [B*S, D]

  float* outp = (float*)d_out;
  float* attn = outp + (size_t)B_ * S_ * D_;

  k_cvt_bf16<<<(B_ * S_ * D_) / 256, 256, 0, stream>>>(x, Xbf, B_ * S_ * D_);
  k_transpose_bf16<<<(D_ * D_) / 256, 256, 0, stream>>>(Wq, WqT, D_, D_);
  k_transpose_bf16<<<(D_ * D_) / 256, 256, 0, stream>>>(Wk, WkT, D_, D_);
  k_transpose_bf16<<<(D_ * D_) / 256, 256, 0, stream>>>(Wv, WvT, D_, D_);
  k_transpose_bf16<<<(D_ * D_) / 256, 256, 0, stream>>>(Wo, WoT, D_, D_);

  k_qkv_proj<<<2048, 256, 0, stream>>>(Xbf, WqT, bq, Qh, 0);
  k_qkv_proj<<<2048, 256, 0, stream>>>(Xbf, WkT, bk, Kh, 0);
  k_qkv_proj<<<2048, 256, 0, stream>>>(Xbf, WvT, bv, Vt, 1);

  size_t smem = (size_t)16 * S_ * 4 + (size_t)16 * S_ * 2 + 288 * 4
              + 16 * 64 * 4 + 16 * 64 * 2;
  k_attn<<<B_ * H_ * (S_ / 16), 256, smem, stream>>>(Qh, Kh, Vt, mask, attn, O);

  k_oproj<<<2048, 256, 0, stream>>>(O, WoT, bo, outp);
}
